// GVPEncoder_33535104647885
// MI455X (gfx1250) — compile-verified, matrix operand order checked by reading, not run on hardware
//
#include <hip/hip_runtime.h>

// ======================= CDNA5 types =======================
typedef __attribute__((ext_vector_type(16))) __bf16 v16bf;
typedef __attribute__((ext_vector_type(8)))  float  v8f;

struct Seg { const float* p; const int* idx; int c; int ld; };

// ======================= device helpers =======================
__device__ __forceinline__ float dot3f(const float* a, const float* b){ return a[0]*b[0]+a[1]*b[1]+a[2]*b[2]; }
__device__ __forceinline__ void cross3f(const float* a, const float* b, float* o){
  o[0]=a[1]*b[2]-a[2]*b[1]; o[1]=a[2]*b[0]-a[0]*b[2]; o[2]=a[0]*b[1]-a[1]*b[0]; }
__device__ __forceinline__ void norm3f(float* v){
  float n = sqrtf(v[0]*v[0]+v[1]*v[1]+v[2]*v[2] + 1e-8f);
  float inv = 1.f/n; v[0]*=inv; v[1]*=inv; v[2]*=inv; }
__device__ __forceinline__ float sigf(float x){ return 1.f/(1.f+expf(-x)); }

// async global->LDS copy (CDNA5): one instruction per wave, per-lane 16B
__device__ __forceinline__ void async_copy16(const __bf16* gsrc, __bf16* ldst){
  unsigned ldsOff = (unsigned)(unsigned long long)(const void*)ldst;
  asm volatile("global_load_async_to_lds_b128 %0, %1, off"
               :: "v"(ldsOff), "v"((unsigned long long)gsrc) : "memory");
}

// ======================= node features =======================
__global__ void k_nodefeat(const float* __restrict__ coords, const unsigned char* __restrict__ cmask,
                           float* __restrict__ nf_s, float* __restrict__ nf_v, int Bb, int Ll)
{
  long t = blockIdx.x*(long)blockDim.x + threadIdx.x;
  long n = (long)Bb*Ll;
  if (t >= n) return;
  int b = (int)(t / Ll), i = (int)(t % Ll);
  const float* Xb = coords + (long)b*Ll*9;   // (3L, 3) flattened atom chain
  int threeL = 3*Ll;
  for (int tt = 0; tt < 3; ++tt) {           // dihedrals (padded 1 front / 2 back)
    int j = i*3 + tt - 1;
    float D = 0.f;
    if (j >= 0 && j < threeL - 3) {
      float u2[3], u1[3], u0[3];
      for (int c=0;c<3;++c){
        u2[c]=Xb[(j+1)*3+c]-Xb[j*3+c];
        u1[c]=Xb[(j+2)*3+c]-Xb[(j+1)*3+c];
        u0[c]=Xb[(j+3)*3+c]-Xb[(j+2)*3+c];
      }
      norm3f(u2); norm3f(u1); norm3f(u0);
      float n2[3], n1[3];
      cross3f(u2,u1,n2); cross3f(u1,u0,n1);
      norm3f(n2); norm3f(n1);
      float cosD = dot3f(n2,n1);
      cosD = fminf(fmaxf(cosD, -1.f+1e-7f), 1.f-1e-7f);
      float sg = dot3f(u2,n1);
      float s = (sg>0.f)?1.f:((sg<0.f)?-1.f:0.f);
      D = s * acosf(cosD);
    }
    nf_s[t*7+tt]   = cosf(D);
    nf_s[t*7+3+tt] = sinf(D);
  }
  nf_s[t*7+6] = cmask[t] ? 1.f : 0.f;
  float fwd[3]={0.f,0.f,0.f}, bwd[3]={0.f,0.f,0.f};
  if (i < Ll-1) { for(int c=0;c<3;++c) fwd[c]=Xb[((i+1)*3+1)*3+c]-Xb[(i*3+1)*3+c]; norm3f(fwd); }
  if (i > 0)    { for(int c=0;c<3;++c) bwd[c]=Xb[((i-1)*3+1)*3+c]-Xb[(i*3+1)*3+c]; norm3f(bwd); }
  float cc[3], nn[3];
  for(int c=0;c<3;++c){
    float o = Xb[(i*3+1)*3+c];
    cc[c]=Xb[(i*3+2)*3+c]-o; nn[c]=Xb[(i*3+0)*3+c]-o;
  }
  norm3f(cc); norm3f(nn);
  float bis[3], perp[3];
  for(int c=0;c<3;++c) bis[c]=cc[c]+nn[c];
  norm3f(bis);
  cross3f(cc,nn,perp); norm3f(perp);
  const float k1 = 0.57735026919f, k2 = 0.81649658092f;
  for(int d=0; d<3; ++d){
    nf_v[(t*3+d)*3+0]=fwd[d];
    nf_v[(t*3+d)*3+1]=bwd[d];
    nf_v[(t*3+d)*3+2]=-bis[d]*k1 - perp[d]*k2;
  }
}

// ======================= KNN: one wave per row, LDS distance cache =======================
__global__ void k_knn(const float* __restrict__ coords, const unsigned char* __restrict__ cmask,
                      const int* __restrict__ res_idx, const unsigned char* __restrict__ pmask,
                      int* __restrict__ e_dst, float* __restrict__ e_dist, int Bb, int Ll, int Kk)
{
  extern __shared__ float sD[];                 // Ll floats
  int row = blockIdx.x;                         // global node 0..B*L-1
  int b = row / Ll, i = row % Ll;
  int lane = threadIdx.x;
  long nb = (long)b*Ll;
  float xi = coords[((nb+i)*3+1)*3+0];
  float yi = coords[((nb+i)*3+1)*3+1];
  float zi = coords[((nb+i)*3+1)*3+2];
  bool cmi = cmask[nb+i] != 0, rmi = pmask[nb+i] == 0;
  int   ri = res_idx[nb+i];
  for (int j = lane; j < Ll; j += 32) {
    float dx = coords[((nb+j)*3+1)*3+0]-xi;
    float dy = coords[((nb+j)*3+1)*3+1]-yi;
    float dz = coords[((nb+j)*3+1)*3+2]-zi;
    float D = sqrtf(dx*dx+dy*dy+dz*dz);
    bool cm2 = cmi && (cmask[nb+j] != 0);
    bool rm2 = rmi && (pmask[nb+j] == 0);
    float Dseq = fabsf((float)(ri - res_idx[nb+j]));
    float Dadj = (cm2 ? D : (1e8f + Dseq*1e6f)) + (rm2 ? 0.f : 1e10f);
    sD[j] = Dadj;
  }
  __syncthreads();
  int chunk = (Ll + 31) / 32;                    // 64 for L=2048
  unsigned long long chosen = 0ull;
  for (int k = 0; k < Kk; ++k) {
    float best = 3e38f; int bidx = -1;
    for (int c = 0; c < chunk; ++c) {
      int j = lane*chunk + c;
      if (j < Ll && !((chosen >> c) & 1ull)) {
        float d = sD[j];
        if (d < best) { best = d; bidx = j; }
      }
    }
    for (int off = 16; off >= 1; off >>= 1) {
      float ob = __shfl_xor(best, off, 32);
      int   oi = __shfl_xor(bidx, off, 32);
      if (ob < best || bidx < 0) { best = ob; bidx = oi; }
    }
    int bsel = __shfl(bidx, 0, 32);
    if (bsel >= 0 && lane == bsel / chunk) chosen |= 1ull << (bsel % chunk);
    if (lane == 0) {
      e_dst[(long)row*Kk + k]  = (bsel >= 0) ? (b*Ll + bsel) : row;
      e_dist[(long)row*Kk + k] = (bsel >= 0) ? sD[bsel] : 0.f;
    }
  }
}

// ======================= edge features =======================
__global__ void k_edge1(const float* __restrict__ coords, const int* __restrict__ e_dst,
                        const float* __restrict__ e_dist, float* __restrict__ e_vec,
                        float* __restrict__ bsum, int Bb, int Ll, int Kk)
{
  long n_e = (long)Bb*Ll*Kk;
  long t = blockIdx.x*(long)blockDim.x + threadIdx.x;
  if (t >= n_e) return;
  long src = t / Kk;
  long dst = e_dst[t];
  float vx = coords[(src*3+1)*3+0] - coords[(dst*3+1)*3+0];
  float vy = coords[(src*3+1)*3+1] - coords[(dst*3+1)*3+1];
  float vz = coords[(src*3+1)*3+2] - coords[(dst*3+1)*3+2];
  e_vec[t*3+0]=vx; e_vec[t*3+1]=vy; e_vec[t*3+2]=vz;
  if (e_dist[t] < 5e7f) {
    int b = (int)(src / Ll);
    atomicAdd(&bsum[b*4+0], vx);
    atomicAdd(&bsum[b*4+1], vy);
    atomicAdd(&bsum[b*4+2], vz);
    atomicAdd(&bsum[b*4+3], 1.f);
  }
}

__global__ void k_edge2(const int* __restrict__ e_dst, const float* __restrict__ e_dist,
                        const float* __restrict__ e_vec, const float* __restrict__ bsum,
                        const int* __restrict__ res_idx, const unsigned char* __restrict__ cmask,
                        float* __restrict__ ef_s, float* __restrict__ ef_v, int Bb, int Ll, int Kk)
{
  long n_e = (long)Bb*Ll*Kk;
  long t = blockIdx.x*(long)blockDim.x + threadIdx.x;
  if (t >= n_e) return;
  long src = t / Kk;
  long dst = e_dst[t];
  int b = (int)(src / Ll);
  float Ed = e_dist[t];
  const float stdr = 20.f/16.f;
  for (int m = 0; m < 16; ++m) {
    float z = (Ed - (float)m*(20.f/15.f)) / stdr;
    ef_s[t*34+m] = expf(-z*z);
  }
  float dd = (float)(res_idx[src] - res_idx[dst]);
  const float base = -9.21034037198f / 16.f;     // -ln(1e4)/16
  for (int m = 0; m < 8; ++m) {
    float freq = expf((float)(2*m) * base);
    float ang = dd * freq;
    ef_s[t*34+16+m] = cosf(ang);
    ef_s[t*34+24+m] = sinf(ang);
  }
  ef_s[t*34+32] = cmask[src] ? 0.f : 1.f;
  ef_s[t*34+33] = cmask[dst] ? 0.f : 1.f;
  float cm = (Ed < 5e7f) ? 1.f : 0.f;
  float cnt = fmaxf(bsum[b*4+3], 1.f);
  float v[3];
  for (int d = 0; d < 3; ++d) {
    float mean = bsum[b*4+d] / cnt;
    v[d] = e_vec[t*3+d]*cm + mean*(1.f-cm);
  }
  norm3f(v);
  ef_v[t*3+0]=v[0]; ef_v[t*3+1]=v[1]; ef_v[t*3+2]=v[2];
}

__global__ void k_buildidx(const int* __restrict__ e_dst, int* __restrict__ srcN,
                           int* __restrict__ srcR3, int* __restrict__ dstR3, long n_e, int Kk)
{
  long t = blockIdx.x*(long)blockDim.x + threadIdx.x;
  if (t >= n_e) return;
  int s = (int)(t / Kk);
  int d = e_dst[t];
  srcN[t] = s;
  for (int j = 0; j < 3; ++j) { srcR3[t*3+j] = s*3+j; dstR3[t*3+j] = d*3+j; }
}

__global__ void k_rbfconf(const float* __restrict__ conf, float* __restrict__ out, long n)
{
  long t = blockIdx.x*(long)blockDim.x + threadIdx.x;
  if (t >= n*16) return;
  long m = t / 16; int j = (int)(t % 16);
  float z = (conf[m] - (float)j*(1.f/15.f)) * 16.f;
  out[t] = expf(-z*z);
}

// ======================= pack (gather + concat + f32->bf16 + pad) =======================
__global__ void k_pack_bf16(__bf16* __restrict__ dst, long Mp, int Kp, long M,
                            Seg s0, Seg s1, Seg s2, Seg s3)
{
  long t = blockIdx.x*(long)blockDim.x + threadIdx.x;
  if (t >= Mp*(long)Kp) return;
  long m = t / Kp; int k = (int)(t % Kp);
  float v = 0.f;
  if (m < M) {
    int kk = k;
    if (kk < s0.c)      { long r = s0.idx ? (long)s0.idx[m] : m; v = s0.p[r*(long)s0.ld + kk]; }
    else { kk -= s0.c;
      if (kk < s1.c)    { long r = s1.idx ? (long)s1.idx[m] : m; v = s1.p[r*(long)s1.ld + kk]; }
      else { kk -= s1.c;
        if (kk < s2.c)  { long r = s2.idx ? (long)s2.idx[m] : m; v = s2.p[r*(long)s2.ld + kk]; }
        else { kk -= s2.c;
          if (kk < s3.c){ long r = s3.idx ? (long)s3.idx[m] : m; v = s3.p[r*(long)s3.ld + kk]; }
        }
      }
    }
  }
  dst[t] = (__bf16)v;
}

// Weight conversion directly into WMMA B-fragment order:
// Wf[(((kt*nTiles)+nt)*32 + lane)*16 + i]  holds W[k][n] with
//   n = nt*16 + lane%16,  k = kt*32 + kpair(i, lane/16)  (ISA 7.12.2 16-bit layout)
__global__ void k_wfrag(__bf16* __restrict__ dst, const float* __restrict__ w,
                        int K, int N, int Kp, int Np)
{
  int t = blockIdx.x*blockDim.x + threadIdx.x;
  if (t >= Kp*Np) return;
  int i    = t & 15;
  int lane = (t >> 4) & 31;
  int tile = t >> 9;                 // kt*nTiles + nt
  int nTiles = Np >> 4;
  int kt = tile / nTiles, nt = tile % nTiles;
  int g = lane >> 4;
  int p = i >> 1;
  int kb = ((p < 4) ? (2*p) : (16 + 2*(p-4))) + 8*g + (i & 1);
  int k = (kt << 5) + kb;
  int n = (nt << 4) + (lane & 15);
  dst[t] = (__bf16)((k < K && n < N) ? w[(long)k*N + n] : 0.f);
}

// ======================= WMMA bf16 GEMM =======================
// C(M,Np) = A(Mp,Kp) @ Wf(Kp,Np frag) [+bias]; 32-row M tile, 64-col N tile,
// double-buffered async global->LDS A staging pipelined against 2 WMMAs/iter.
__launch_bounds__(128)
__global__ void k_gemm_bf16(const __bf16* __restrict__ A, const __bf16* __restrict__ Wf,
                            const float* __restrict__ bias, float* __restrict__ C,
                            int M, int Np, int Kp, int ldc, int addC)
{
  __shared__ __align__(16) __bf16 lA[2][32*32];  // 2 x 2KB A tiles
  int tid  = threadIdx.x;
  int lane = tid & 31;
  int wave = tid >> 5;
  long row0  = (long)blockIdx.x * 32;
  int colBlk = blockIdx.y * 64;
  int col0   = colBlk + wave*16;
  int nTiles = Np >> 4;
  int nt = col0 >> 4; if (nt >= nTiles) nt = nTiles - 1;   // clamp; store is guarded below
  int kTiles = Kp >> 5;
  v8f acc0 = {0.f,0.f,0.f,0.f,0.f,0.f,0.f,0.f};
  v8f acc1 = {0.f,0.f,0.f,0.f,0.f,0.f,0.f,0.f};
  int sr = tid >> 2, sch = tid & 3;              // 128 threads stage 32 rows x 16B
  // prologue: async-issue tile 0 into buffer 0 (one async instr per wave)
  async_copy16(A + (row0 + sr)*(long)Kp + sch*8, &lA[0][sr*32 + sch*8]);
  for (int kt = 0; kt < kTiles; ++kt) {
    int cur = kt & 1;
    if (kt + 1 < kTiles) {                       // overlap DMA of tile kt+1 with compute of kt
      async_copy16(A + (row0 + sr)*(long)Kp + (kt+1)*32 + sch*8, &lA[cur^1][sr*32 + sch*8]);
      asm volatile("s_wait_asynccnt 0x1" ::: "memory");   // tile kt landed; kt+1 in flight
      __builtin_prefetch(Wf + (((long)(kt+1)*nTiles + nt)*32 + lane)*16, 0, 3);
    } else {
      asm volatile("s_wait_asynccnt 0x0" ::: "memory");
    }
    // per-lane B fragment: 32 contiguous bytes -> two b128 loads, no guards
    union { v16bf v; uint4 q[2]; } fb;
    const uint4* wsrc = (const uint4*)(Wf + (((long)kt*nTiles + nt)*32 + lane)*16);
    fb.q[0] = wsrc[0];
    fb.q[1] = wsrc[1];
    __syncthreads();
    // A fragments (ISA 7.12.2): per lane two contiguous 16B runs -> ds_load_b128 x2 each
    int mRow = lane & 15, g = lane >> 4;
    const uint4* lq = (const uint4*)&lA[cur][0];   // 16B units within current tile
    union { v16bf v; uint4 q[2]; } fa0, fa1;
    fa0.q[0] = lq[mRow*4 + g];
    fa0.q[1] = lq[mRow*4 + 2 + g];
    fa1.q[0] = lq[(mRow+16)*4 + g];
    fa1.q[1] = lq[(mRow+16)*4 + 2 + g];
    acc0 = __builtin_amdgcn_wmma_f32_16x16x32_bf16(false, fa0.v, false, fb.v,
                                                   (short)0, acc0, false, false);
    acc1 = __builtin_amdgcn_wmma_f32_16x16x32_bf16(false, fa1.v, false, fb.v,
                                                   (short)0, acc1, false, false);
    __syncthreads();                              // reads done before buffer reuse
  }
  if (col0 < Np) {
    int nCol = lane & 15, g = lane >> 4;
    int col = col0 + nCol;
    float bsum = bias ? bias[col] : 0.f;
    #pragma unroll
    for (int r = 0; r < 8; ++r) {
      long rowA = row0 + r + 8*g;                 // C layout: M = r + 8*(lane/16)
      if (rowA < M) {
        float v = acc0[r] + bsum;
        float* p = C + rowA*(long)ldc + col;
        if (addC) *p += v; else *p = v;
      }
      long rowB = row0 + 16 + r + 8*g;
      if (rowB < M) {
        float v = acc1[r] + bsum;
        float* p = C + rowB*(long)ldc + col;
        if (addC) *p += v; else *p = v;
      }
    }
  }
}

// ======================= elementwise =======================
__global__ void k_vnorm(float* __restrict__ vn, const float* __restrict__ vhT,
                        long n, int h, int ldh, int ldvn)
{
  long t = blockIdx.x*(long)blockDim.x + threadIdx.x;
  if (t >= n*(long)h) return;
  long m = t / h; int c = (int)(t % h);
  float a = vhT[(m*3+0)*(long)ldh + c];
  float b = vhT[(m*3+1)*(long)ldh + c];
  float d = vhT[(m*3+2)*(long)ldh + c];
  vn[m*(long)ldvn + c] = sqrtf(fmaxf(a*a+b*b+d*d, 1e-8f));
}

__global__ void k_act(float* __restrict__ dst, const float* __restrict__ src, long total, int mode)
{
  long t = blockIdx.x*(long)blockDim.x + threadIdx.x;
  if (t >= total) return;
  float v = src[t];
  dst[t] = (mode == 1) ? fmaxf(v, 0.f) : sigf(v);
}

__global__ void k_gate(float* __restrict__ vT, const float* __restrict__ gz,
                       long n, int nv, int ldv, int ldg)
{
  long t = blockIdx.x*(long)blockDim.x + threadIdx.x;
  if (t >= n*3*(long)nv) return;
  long m = t / (3*nv); int r = (int)(t % (3*nv));
  int d = r / nv, c = r % nv;
  vT[(m*3+d)*(long)ldv + c] *= sigf(gz[m*(long)ldg + c]);
}

__global__ void k_gvpln(float* __restrict__ s, float* __restrict__ vT,
                        const float* __restrict__ g, const float* __restrict__ b,
                        long n, int ns, int nv, int ldv)
{
  long m = blockIdx.x*(long)blockDim.x + threadIdx.x;
  if (m >= n) return;
  const float eps = 1e-4f;
  float sumvn = 0.f, sumnz = 0.f;
  for (int c = 0; c < nv; ++c) {
    float v0 = vT[(m*3+0)*(long)ldv+c], v1 = vT[(m*3+1)*(long)ldv+c], v2 = vT[(m*3+2)*(long)ldv+c];
    float vn = fmaxf(v0*v0+v1*v1+v2*v2, eps);
    float nz = (vn > 2.f*eps) ? 1.f : 0.f;
    sumvn += vn*nz; sumnz += nz;
  }
  float vm  = sumvn / (eps + sumnz);
  float inv = 1.f / sqrtf(vm + eps);
  for (int c = 0; c < nv; ++c) {
    float v0 = vT[(m*3+0)*(long)ldv+c], v1 = vT[(m*3+1)*(long)ldv+c], v2 = vT[(m*3+2)*(long)ldv+c];
    float vn = fmaxf(v0*v0+v1*v1+v2*v2, eps);
    float nz = (vn > 2.f*eps) ? 1.f : 0.f;
    vT[(m*3+0)*(long)ldv+c] = nz*v0*inv;
    vT[(m*3+1)*(long)ldv+c] = nz*v1*inv;
    vT[(m*3+2)*(long)ldv+c] = nz*v2*inv;
  }
  float mu = 0.f;
  for (int c = 0; c < ns; ++c) mu += s[m*(long)ns+c];
  mu /= (float)ns;
  float var = 0.f;
  for (int c = 0; c < ns; ++c) { float d0 = s[m*(long)ns+c]-mu; var += d0*d0; }
  var /= (float)ns;
  float si = 1.f / sqrtf(var + 1e-5f);
  for (int c = 0; c < ns; ++c)
    s[m*(long)ns+c] = (s[m*(long)ns+c]-mu)*si*g[c] + b[c];
}

__global__ void k_scatter(float* __restrict__ agg_s, float* __restrict__ agg_v, float* __restrict__ cnt,
                          const float* __restrict__ ms, const float* __restrict__ mvT,
                          const int* __restrict__ dst, long n_e, int ns, int nv, int ldms, int ldmv)
{
  long per = ns + 3*nv + 1;
  long t = blockIdx.x*(long)blockDim.x + threadIdx.x;
  if (t >= n_e*per) return;
  long e = t / per; int j = (int)(t % per);
  int d0 = dst[e];
  if (j < ns) atomicAdd(&agg_s[(long)d0*ns + j], ms[e*(long)ldms + j]);
  else if (j < ns + 3*nv) {
    int r = j - ns; int d = r / nv, c = r % nv;
    atomicAdd(&agg_v[((long)d0*3 + d)*nv + c], mvT[(e*3+d)*(long)ldmv + c]);
  } else atomicAdd(&cnt[d0], 1.f);
}

__global__ void k_residual(float* __restrict__ outS, float* __restrict__ outV,
                           const float* __restrict__ s, const float* __restrict__ vT,
                           const float* __restrict__ ds, const float* __restrict__ dv,
                           const float* __restrict__ cnt, long n, int ns, int nv, int ldds, int lddv)
{
  long per = ns + 3*nv;
  long t = blockIdx.x*(long)blockDim.x + threadIdx.x;
  if (t >= n*per) return;
  long m = t / per; int j = (int)(t % per);
  float c = cnt ? fmaxf(cnt[m], 1.f) : 1.f;
  if (j < ns) outS[m*(long)ns+j] = s[m*(long)ns+j] + ds[m*(long)ldds+j]/c;
  else {
    int r = j - ns; int d = r / nv, ch = r % nv;
    outV[(m*3+d)*(long)nv+ch] = vT[(m*3+d)*(long)nv+ch] + dv[(m*3+d)*(long)lddv+ch]/c;
  }
}

__global__ void k_writeout(const float* __restrict__ hs, const float* __restrict__ hvT,
                           float* __restrict__ out, long n, int ns, int nv)
{
  long t = blockIdx.x*(long)blockDim.x + threadIdx.x;
  long tot = n*(ns + nv*3);
  if (t >= tot) return;
  if (t < n*(long)ns) out[t] = hs[t];
  else {
    long r = t - n*(long)ns;
    long m = r / (nv*3); int rem = (int)(r % (nv*3));
    int c = rem / 3, d = rem % 3;
    out[t] = hvT[(m*3+d)*(long)nv + c];      // (n,NV,3) from transposed (n,3,NV)
  }
}

// ======================= host =======================
static inline long ceil32l(long x){ return (x+31)/32*32; }
static inline int  ceil32i(int x){ return (x+31)/32*32; }
static inline int  ceil16i(int x){ return (x+15)/16*16; }

extern "C" void kernel_launch(void* const* d_in, const int* in_sizes, int n_in,
                              void* d_out, int out_size, void* d_ws, size_t ws_size,
                              hipStream_t stream)
{
  (void)in_sizes; (void)out_size; (void)ws_size;
  if (n_in < 109) return;
  const int Bb = 4, Ll = 2048, Kk = 30, NSc = 128, NVc = 32, ESc = 32;
  const long n   = (long)Bb*Ll;          // 8192
  const long n_e = n*Kk;                 // 245760
  const long n3  = n*3, ne3 = n_e*3;

  auto F = [&](int i){ return (const float*)d_in[i]; };
  const float* coords = F(0);
  const unsigned char* cmask = (const unsigned char*)d_in[1];
  const int* res_idx = (const int*)d_in[2];
  const unsigned char* pmask = (const unsigned char*)d_in[3];
  const float* conf = F(4);
  // params: sorted-key pytree flatten, base 5
  const int P = 5;
  const int CONF_B=P+0, CONF_W=P+1, ELN_B=P+2, ELN_G=P+3;
  const int EE_WH=P+4, EE_WSB=P+5, EE_WSW=P+6, EE_WV=P+7;
  const int EN_WH=P+8, EN_WSB=P+9, EN_WSW=P+10, EN_WV=P+11;
  const int LYR0 = P+12, LSTR = 30;
  const int NLN_B = P+102, NLN_G = P+103;

  // ---- workspace bump allocator ----
  char* base = (char*)d_ws; size_t off = 0;
  auto alloc = [&](size_t bytes)->void* {
    void* p = base + off; off += (bytes + 255) & ~(size_t)255; return p; };
  float* nf_s  = (float*)alloc(n*7*4);
  float* nf_v  = (float*)alloc(n*9*4);
  float* rbfb  = (float*)alloc(n*16*4);
  int*   e_dst = (int*)  alloc(n_e*4);
  float* e_dis = (float*)alloc(n_e*4);
  float* e_vec = (float*)alloc(n_e*3*4);
  float* bsum  = (float*)alloc(Bb*4*4);
  float* ef_s  = (float*)alloc(n_e*34*4);
  float* ef_v  = (float*)alloc(n_e*3*4);
  int*   srcN  = (int*)  alloc(n_e*4);
  int*   srcR3 = (int*)  alloc(ne3*4);
  int*   dstR3 = (int*)  alloc(ne3*4);
  float* hs    = (float*)alloc(n*128*4);
  float* hvT   = (float*)alloc(n*96*4);
  float* hs2   = (float*)alloc(n*128*4);
  float* hvT2  = (float*)alloc(n*96*4);
  float* es    = (float*)alloc(n_e*32*4);
  float* evT   = (float*)alloc(ne3*16*4);
  float* vnb   = (float*)alloc(n_e*80*4);
  float* vhT   = (float*)alloc(ne3*80*4);
  float* msA   = (float*)alloc(n_e*128*4);
  float* msB   = (float*)alloc(n_e*128*4);
  float* mvA   = (float*)alloc(ne3*32*4);
  float* mvB   = (float*)alloc(ne3*32*4);
  float* sigb  = (float*)alloc(n_e*128*4);
  float* gzb   = (float*)alloc(n_e*32*4);
  float* agg_s = (float*)alloc(n*128*4);
  float* agg_v = (float*)alloc(n*96*4);
  float* cntb  = (float*)alloc(n*4);
  float* fsb   = (float*)alloc(n*512*4);
  float* fvb   = (float*)alloc(n*192*4);
  float* fs2   = (float*)alloc(n*128*4);
  float* fv2   = (float*)alloc(n*96*4);
  __bf16* packA = (__bf16*)alloc((size_t)n_e*384*2);
  __bf16* packB = (__bf16*)alloc((size_t)ne3*96*2);
  __bf16* wbuf  = (__bf16*)alloc((size_t)131072*2);

  auto grid1 = [&](long t){ return dim3((unsigned)((t + 255)/256)); };
  Seg Z{nullptr,nullptr,0,0};

  auto pack = [&](__bf16* dst, long M, int Kp, Seg a, Seg b, Seg c, Seg d){
    long Mp = ceil32l(M);                       // pad to 32-row GEMM tiles
    k_pack_bf16<<<grid1(Mp*(long)Kp),256,0,stream>>>(dst, Mp, Kp, M, a, b, c, d);
  };
  auto gemm = [&](const __bf16* A, long M, int Kp, const float* wF, int K, int N, int Np,
                  const float* biasF, float* C, int ldc, int addC){
    k_wfrag<<<grid1((long)Kp*Np),256,0,stream>>>(wbuf, wF, K, N, Kp, Np);
    dim3 g((unsigned)(ceil32l(M)/32), (unsigned)((Np+63)/64));
    k_gemm_bf16<<<g,128,0,stream>>>(A, wbuf, biasF, C, (int)M, Np, Kp, ldc, addC);
  };
  auto act = [&](float* dst, const float* src, long tot, int mode){
    k_act<<<grid1(tot),256,0,stream>>>(dst, src, tot, mode);
  };
  // generic GVP after the caller has packed mvT into packA (rows3 x ceil32(vi))
  auto gvp_core = [&](long rows, int vi, int h, int so, int vo,
                      Seg s0, Seg s1, Seg s2,
                      const float* wh, const float* wsb, const float* wsw,
                      const float* wsvb, const float* wsvw, const float* wv,
                      bool gate, bool relu_s,
                      float* outS, float* outV){
    long rows3 = rows*3;
    int KpV = ceil32i(vi), Nph = ceil16i(h);
    gemm(packA, rows3, KpV, wh, vi, h, Nph, nullptr, vhT, Nph, 0);
    k_vnorm<<<grid1(rows*(long)h),256,0,stream>>>(vnb, vhT, rows, h, Nph, Nph);
    int Ks = s0.c + s1.c + s2.c + h, KpS = ceil32i(Ks);
    Seg sv{vnb, nullptr, h, Nph};
    pack(packA, rows, KpS, s0, s1, s2, sv);
    gemm(packA, rows, KpS, wsw, Ks, so, so, wsb, outS, so, 0);
    int KpH = ceil32i(h), Npv = ceil16i(vo);
    Seg svh{vhT, nullptr, h, Nph};
    pack(packB, rows3, KpH, svh, Z, Z, Z);
    gemm(packB, rows3, KpH, wv, h, vo, Npv, nullptr, outV, Npv, 0);
    if (gate) {
      act(sigb, outS, rows*(long)so, 2);
      Seg sg{sigb, nullptr, so, so};
      pack(packB, rows, ceil32i(so), sg, Z, Z, Z);
      gemm(packB, rows, ceil32i(so), wsvw, so, vo, Npv, wsvb, gzb, Npv, 0);
      k_gate<<<grid1(rows*3*(long)vo),256,0,stream>>>(outV, gzb, rows, vo, Npv, Npv);
    }
    if (relu_s) act(outS, outS, rows*(long)so, 1);
  };

  // ================= features =================
  k_nodefeat<<<grid1(n),256,0,stream>>>(coords, cmask, nf_s, nf_v, Bb, Ll);
  k_knn<<<dim3((unsigned)n),32,Ll*sizeof(float),stream>>>(coords, cmask, res_idx, pmask,
                                                          e_dst, e_dis, Bb, Ll, Kk);
  hipMemsetAsync(bsum, 0, Bb*4*4, stream);
  k_edge1<<<grid1(n_e),256,0,stream>>>(coords, e_dst, e_dis, e_vec, bsum, Bb, Ll, Kk);
  k_edge2<<<grid1(n_e),256,0,stream>>>(e_dst, e_dis, e_vec, bsum, res_idx, cmask,
                                       ef_s, ef_v, Bb, Ll, Kk);
  k_buildidx<<<grid1(n_e),256,0,stream>>>(e_dst, srcN, srcR3, dstR3, n_e, Kk);

  // ================= node embedding =================
  { Seg sv{nf_v, nullptr, 3, 3};
    pack(packA, n3, 32, sv, Z, Z, Z);
    Seg ss{nf_s, nullptr, 7, 7};
    gvp_core(n, 3, 32, 128, 32, ss, Z, Z,
             F(EN_WH), F(EN_WSB), F(EN_WSW), nullptr, nullptr, F(EN_WV),
             false, false, hs, hvT);
  }
  k_gvpln<<<grid1(n),256,0,stream>>>(hs, hvT, F(NLN_G), F(NLN_B), n, 128, 32, 32);
  k_rbfconf<<<grid1(n*16),256,0,stream>>>(conf, rbfb, n);
  { Seg sr{rbfb, nullptr, 16, 16};
    pack(packB, n, 32, sr, Z, Z, Z);
    gemm(packB, n, 32, F(CONF_W), 16, 128, 128, F(CONF_B), hs, 128, 1); }

  // ================= edge embedding =================
  { Seg sv{ef_v, nullptr, 1, 1};
    pack(packA, ne3, 32, sv, Z, Z, Z);
    Seg ss{ef_s, nullptr, 34, 34};
    gvp_core(n_e, 1, 1, 32, 1, ss, Z, Z,
             F(EE_WH), F(EE_WSB), F(EE_WSW), nullptr, nullptr, F(EE_WV),
             false, false, es, evT);           // evT ld = 16
  }
  k_gvpln<<<grid1(n_e),256,0,stream>>>(es, evT, F(ELN_G), F(ELN_B), n_e, 32, 1, 16);

  // ================= conv layers =================
  for (int l = 0; l < 3; ++l) {
    int LB = LYR0 + l*LSTR;
    int FF0=LB+0, FF1=LB+6, M0=LB+10, M1=LB+16, M2=LB+22, N0B=LB+26, N0G=LB+27, N1B=LB+28, N1G=LB+29;
    // --- m0 ---
    { Seg a{hvT, srcR3, 32, 32}, b{evT, nullptr, 1, 16}, c{hvT, dstR3, 32, 32};
      pack(packA, ne3, 96, a, b, c, Z);
      Seg s0{hs, srcN, 128, 128}, s1{es, nullptr, 32, 32}, s2{hs, e_dst, 128, 128};
      gvp_core(n_e, 65, 65, 128, 32, s0, s1, s2,
               F(M0+0), F(M0+1), F(M0+2), F(M0+3), F(M0+4), F(M0+5),
               true, true, msA, mvA); }
    // --- m1 ---
    { Seg a{mvA, nullptr, 32, 32};
      pack(packA, ne3, 32, a, Z, Z, Z);
      Seg s0{msA, nullptr, 128, 128};
      gvp_core(n_e, 32, 32, 128, 32, s0, Z, Z,
               F(M1+0), F(M1+1), F(M1+2), F(M1+3), F(M1+4), F(M1+5),
               true, true, msB, mvB); }
    // --- m2 (no gate / act) ---
    { Seg a{mvB, nullptr, 32, 32};
      pack(packA, ne3, 32, a, Z, Z, Z);
      Seg s0{msB, nullptr, 128, 128};
      gvp_core(n_e, 32, 32, 128, 32, s0, Z, Z,
               F(M2+0), F(M2+1), F(M2+2), nullptr, nullptr, F(M2+3),
               false, false, msA, mvA); }
    // --- aggregate (segment mean over dst) + residual + norm0 ---
    hipMemsetAsync(agg_s, 0, n*128*4, stream);
    hipMemsetAsync(agg_v, 0, n*96*4, stream);
    hipMemsetAsync(cntb, 0, n*4, stream);
    k_scatter<<<grid1(n_e*(long)(128+96+1)),256,0,stream>>>(agg_s, agg_v, cntb, msA, mvA,
                                                            e_dst, n_e, 128, 32, 128, 32);
    k_residual<<<grid1(n*(long)(128+96)),256,0,stream>>>(hs2, hvT2, hs, hvT, agg_s, agg_v,
                                                         cntb, n, 128, 32, 128, 32);
    k_gvpln<<<grid1(n),256,0,stream>>>(hs2, hvT2, F(N0G), F(N0B), n, 128, 32, 32);
    // --- ff0 ---
    { Seg a{hvT2, nullptr, 32, 32};
      pack(packA, n3, 32, a, Z, Z, Z);
      Seg s0{hs2, nullptr, 128, 128};
      gvp_core(n, 32, 64, 512, 64, s0, Z, Z,
               F(FF0+0), F(FF0+1), F(FF0+2), F(FF0+3), F(FF0+4), F(FF0+5),
               true, true, fsb, fvb); }
    // --- ff1 ---
    { Seg a{fvb, nullptr, 64, 64};
      pack(packA, n3, 64, a, Z, Z, Z);
      Seg s0{fsb, nullptr, 512, 512};
      gvp_core(n, 64, 64, 128, 32, s0, Z, Z,
               F(FF1+0), F(FF1+1), F(FF1+2), nullptr, nullptr, F(FF1+3),
               false, false, fs2, fv2); }
    // --- residual + norm1 (back into hs / hvT) ---
    k_residual<<<grid1(n*(long)(128+96)),256,0,stream>>>(hs, hvT, hs2, hvT2, fs2, fv2,
                                                         nullptr, n, 128, 32, 128, 32);
    k_gvpln<<<grid1(n),256,0,stream>>>(hs, hvT, F(N1G), F(N1B), n, 128, 32, 32);
  }

  // ================= output =================
  k_writeout<<<grid1(n*(long)(NSc + NVc*3)),256,0,stream>>>(hs, hvT, (float*)d_out, n, NSc, NVc);
  (void)ESc;
}
